// GenePredHMMEmitter_48558900249192
// MI455X (gfx1250) — matrix operand order repaired and checked by reading, not compiled, not run on hardware
//
#include <hip/hip_runtime.h>

typedef __attribute__((ext_vector_type(16))) _Float16 v16h;
typedef __attribute__((ext_vector_type(8)))  float    v8f;

#define L_SEQ   8192
#define B_BATCH 16
#define N_TILES ((B_BATCH * L_SEQ) / 16)   // 8192 tiles of 16 positions

// Padded LDS strides (in halves) -> conflict-free row access across 16 lanes
#define SW_STRIDE 136   // 128 + 8
#define SB_STRIDE 40    // 32 + 8
#define SC_STRIDE 40
#define SF_STRIDE 136
#define S3_STRIDE 72    // 64 + 8

__device__ __forceinline__ float softplusf(float x) {
  return (x > 20.f) ? x : log1pf(__expf(x));
}

// ---------------------------------------------------------------------------
// Prep kernel: build softmax(B), MVN weight matrix W (f16), per-state K_q,
// and shifted LEFT/RIGHT codon constant matrices in workspace.
// ---------------------------------------------------------------------------
__global__ void genehmm_prep_kernel(const float* __restrict__ ek,   // (15,15)
                                    const float* __restrict__ ee,   // (15,128)
                                    _Float16* __restrict__ wsW,     // (16,128)
                                    float*    __restrict__ wsKq,    // (16)
                                    _Float16* __restrict__ wsBsm,   // (16,32)
                                    _Float16* __restrict__ wsL,     // (16,64)
                                    _Float16* __restrict__ wsR) {   // (16,64)
  const int t = threadIdx.x;
  if (t < 16) wsKq[t] = 0.f;
  __syncthreads();

  // --- softmax of emission kernel rows, zero-padded to 16x32 ---
  if (t < 16) {
    if (t < 15) {
      float m = -1e30f;
      for (int s = 0; s < 15; ++s) m = fmaxf(m, ek[t * 15 + s]);
      float e[15], sum = 0.f;
      for (int s = 0; s < 15; ++s) { e[s] = __expf(ek[t * 15 + s] - m); sum += e[s]; }
      float inv = 1.f / sum;
      for (int s = 0; s < 15; ++s) wsBsm[t * 32 + s] = (_Float16)(e[s] * inv);
      for (int s = 15; s < 32; ++s) wsBsm[t * 32 + s] = (_Float16)0.f;
    } else {
      for (int s = 0; s < 32; ++s) wsBsm[15 * 32 + s] = (_Float16)0.f;
    }
  }

  // --- MVN: logpdf = f.W_q + K_q  with f = [x, x^2] (dim 128) ---
  const float basec = logf(expm1f(sqrtf(0.05f)));   // INIT_VAR = 0.05
  for (int idx = t; idx < 16 * 64; idx += blockDim.x) {
    const int q = idx >> 6, d = idx & 63;
    if (q < 15) {
      const float mu = ee[q * 128 + d];
      const float sr = ee[q * 128 + 64 + d];
      const float sd = softplusf(sr + basec);
      const float i2 = 1.f / (sd * sd);
      float c1 = fminf(fmaxf(mu * i2, -60000.f), 60000.f);   // coeff of x_d
      float c2 = fmaxf(-0.5f * i2, -60000.f);                // coeff of x_d^2
      wsW[q * 128 + d]      = (_Float16)c1;
      wsW[q * 128 + 64 + d] = (_Float16)c2;
      atomicAdd(&wsKq[q], -0.5f * mu * mu * i2 - logf(sd));
    } else {
      wsW[q * 128 + d]      = (_Float16)0.f;
      wsW[q * 128 + 64 + d] = (_Float16)0.f;
    }
  }
  __syncthreads();
  if (t < 15) wsKq[t] -= 58.8120664f;   // 0.5 * 64 * ln(2*pi)

  // --- LEFT/RIGHT codon matrices, shifted so row q corresponds to state q ---
  for (int idx = t; idx < 16 * 64; idx += blockDim.x) {
    const int q = idx >> 6, d = idx & 63;
    const int j = q - 6;                // codon row index
    float lv = 0.f, rv = 0.f;
    if (j >= 0 && j < 9) {
      // LEFT: [ANY, START, IB, IB, IB, ANY, ANY, ANY, ANY]
      if (j == 1)                lv = (d == 14) ? 1.f : 0.f;            // ATG
      else if (j >= 2 && j <= 4) lv = ((d & 15) == 11) ? 0.25f : 0.f;   // NGT
      else                       lv = 0.015625f;                        // NNN
      // RIGHT: [NOT_STOP, ANY, ANY, NOT_STOP, ANY, IEP, IEP, IEP, STOP]
      const bool stop = (d == 48) || (d == 50) || (d == 56);
      if (j == 0 || j == 3)           rv = stop ? 0.f : (1.f / 61.f);
      else if (j == 1 || j == 2 || j == 4) rv = 0.015625f;
      else if (j >= 5 && j <= 7)      rv = (d >= 8 && d < 12) ? 0.25f : 0.f; // AGN
      else  rv = (d == 48) ? 0.34f : ((d == 50) ? 0.33f : ((d == 56) ? 0.33f : 0.f));
    }
    wsL[q * 64 + d] = (_Float16)lv;
    wsR[q * 64 + d] = (_Float16)rv;
  }
}

// ---------------------------------------------------------------------------
// Fragment assembly from LDS (wave32), contiguous K order used for both A and B.
// A 16x32: lane holds row M=lane&15; lanes<16 K=0..15, lanes>=16 K=16..31.
// B 32x16: lane holds col N=lane&15; lanes<16 K=0..15, lanes>=16 K=16..31.
// ---------------------------------------------------------------------------
union FragU { v16h v; unsigned u[8]; };

__device__ __forceinline__ v16h load_fragA(const _Float16* mat, int stride,
                                           int m, int hi, int kb) {
  FragU f;
  const int koff = hi ? 16 : 0;
  const char* base = (const char*)(mat + m * stride + kb + koff);
#pragma unroll
  for (int v = 0; v < 8; ++v) {
    f.u[v]     = *(const unsigned*)(base + 4 * v);        // K = koff+2v
  }
  return f.v;
}

__device__ __forceinline__ v16h load_fragB(const _Float16* mat, int stride,
                                           int n, int hi, int kb) {
  FragU f;
  const int koff = hi ? 16 : 0;
  const char* base = (const char*)(mat + n * stride + kb + koff);
#pragma unroll
  for (int v = 0; v < 8; ++v)
    f.u[v] = *(const unsigned*)(base + 4 * v);            // K = koff+2v
  return f.v;
}

// ---------------------------------------------------------------------------
// Main kernel: 4 waves per block, each wave owns one 16-position tile.
// ---------------------------------------------------------------------------
__global__ __launch_bounds__(128) void genehmm_emitter_kernel(
    const float* __restrict__ in,        // (16,8192,84)
    const float* __restrict__ eh,        // (16,2,15)
    const _Float16* __restrict__ wsW, const float* __restrict__ wsKq,
    const _Float16* __restrict__ wsBsm, const _Float16* __restrict__ wsL,
    const _Float16* __restrict__ wsR,
    float* __restrict__ out) {           // (16,8192,15)
  __shared__ __align__(16) _Float16 sW[16 * SW_STRIDE];
  __shared__ __align__(16) _Float16 sB[16 * SB_STRIDE];
  __shared__ __align__(16) _Float16 sLm[16 * S3_STRIDE];
  __shared__ __align__(16) _Float16 sRm[16 * S3_STRIDE];
  __shared__ float sKq[16];
  __shared__ __align__(16) _Float16 sCls[4][16 * SC_STRIDE];
  __shared__ __align__(16) _Float16 sFeat[4][16 * SF_STRIDE];
  __shared__ __align__(16) _Float16 sL3[4][16 * S3_STRIDE];
  __shared__ __align__(16) _Float16 sR3[4][16 * S3_STRIDE];
  __shared__ float sN[4][80];           // 20 nucleotide 4-vectors per wave

  const int tid = threadIdx.x;
  // stage constants (shared by the whole block)
  for (int i = tid; i < 16 * 128; i += 128) sW[(i >> 7) * SW_STRIDE + (i & 127)] = wsW[i];
  for (int i = tid; i < 16 * 32;  i += 128) sB[(i >> 5) * SB_STRIDE + (i & 31)]  = wsBsm[i];
  for (int i = tid; i < 16 * 64;  i += 128) {
    const int r = i >> 6, c = i & 63;
    sLm[r * S3_STRIDE + c] = wsL[i];
    sRm[r * S3_STRIDE + c] = wsR[i];
  }
  if (tid < 16) sKq[tid] = wsKq[tid];
  __syncthreads();

  const int wave = tid >> 5, lane = tid & 31;
  const int p = lane & 15, hi = lane >> 4;
  _Float16* cls  = sCls[wave];
  _Float16* feat = sFeat[wave];
  _Float16* l3f  = sL3[wave];
  _Float16* r3f  = sR3[wave];
  float*    narr = sN[wave];

  const int gw = blockIdx.x * 4 + wave;
  const int nw = gridDim.x * 4;
  for (int t = gw; t < N_TILES; t += nw) {
    const int b  = t >> 9;               // 512 tiles per sequence
    const int l0 = (t & 511) << 4;
    const float* base = in + (size_t)(b * L_SEQ + l0) * 84;

    // --- stage class probs + embedding features ([x, x^2]) ---
    for (int pp = 0; pp < 16; ++pp) {
      const float* row = base + pp * 84;
      const float cv = (lane < 15) ? row[lane] : 0.f;
      cls[pp * SC_STRIDE + lane] = (_Float16)cv;
      const float x0 = row[15 + lane];        // d = lane
      const float x1 = row[47 + lane];        // d = 32 + lane
      feat[pp * SF_STRIDE + lane]      = (_Float16)x0;
      feat[pp * SF_STRIDE + 64 + lane] = (_Float16)(x0 * x0);
      feat[pp * SF_STRIDE + 32 + lane] = (_Float16)x1;
      feat[pp * SF_STRIDE + 96 + lane] = (_Float16)(x1 * x1);
    }

    // --- nucleotide 4-vectors for l0-2 .. l0+17 (0.25 pad off-sequence) ---
    for (int idx = lane; idx < 80; idx += 32) {
      const int row = idx >> 2, i = idx & 3;
      const int l = l0 + row - 2;
      float nv = 0.25f;
      if (l >= 0 && l < L_SEQ) {
        const float* r = in + (size_t)(b * L_SEQ + l) * 84 + 79;
        nv = r[i] + 0.25f * r[4];
      }
      narr[idx] = nv;
    }

    // --- 3-mer outer products: lanes<16 build l3, lanes>=16 build r3 ---
    {
      const int r0 = hi ? p : (p + 2);           // r3: rows p..p+2, l3: p+2..p+4
      const float* a0 = narr + r0 * 4;
      const float* a1 = narr + (r0 + 1) * 4;
      const float* a2 = narr + (r0 + 2) * 4;
      _Float16* dst = hi ? r3f : l3f;
#pragma unroll
      for (int i = 0; i < 4; ++i) {
        const float vi = a0[i];
#pragma unroll
        for (int j = 0; j < 4; ++j) {
          const float vij = vi * a1[j];
#pragma unroll
          for (int k = 0; k < 4; ++k)
            dst[p * S3_STRIDE + i * 16 + j * 4 + k] = (_Float16)(vij * a2[k]);
        }
      }
    }

    // --- WMMA: MVN GEMM (K=128), class GEMM (K=32), codon GEMMs (K=64 x2) ---
    v8f accM = (v8f)0.0f;
#pragma unroll
    for (int kk = 0; kk < 4; ++kk) {
      const v16h a  = load_fragA(sW,  SW_STRIDE, p, hi, kk * 32);
      const v16h bb = load_fragB(feat, SF_STRIDE, p, hi, kk * 32);
      accM = __builtin_amdgcn_wmma_f32_16x16x32_f16(false, a, false, bb,
                                                    (short)0, accM, false, false);
    }
    v8f accC = (v8f)0.0f;
    {
      const v16h a  = load_fragA(sB,  SB_STRIDE, p, hi, 0);
      const v16h bb = load_fragB(cls, SC_STRIDE, p, hi, 0);
      accC = __builtin_amdgcn_wmma_f32_16x16x32_f16(false, a, false, bb,
                                                    (short)0, accC, false, false);
    }
    v8f accL = (v8f)0.0f, accR = (v8f)0.0f;
#pragma unroll
    for (int kk = 0; kk < 2; ++kk) {
      const v16h al = load_fragA(sLm, S3_STRIDE, p, hi, kk * 32);
      const v16h bl = load_fragB(l3f, S3_STRIDE, p, hi, kk * 32);
      accL = __builtin_amdgcn_wmma_f32_16x16x32_f16(false, al, false, bl,
                                                    (short)0, accL, false, false);
      const v16h ar = load_fragA(sRm, S3_STRIDE, p, hi, kk * 32);
      const v16h br = load_fragB(r3f, S3_STRIDE, p, hi, kk * 32);
      accR = __builtin_amdgcn_wmma_f32_16x16x32_f16(false, ar, false, br,
                                                    (short)0, accR, false, false);
    }

    // --- epilogue: lane holds (q = r + 8*hi, position p) ---
    const int l = l0 + p;
    float* orow = out + ((size_t)b * L_SEQ + (size_t)l) * 15;
#pragma unroll
    for (int r = 0; r < 8; ++r) {
      const int q = r + 8 * hi;
      if (q < 15) {
        const float emb  = __expf((accM[r] + sKq[q]) * 0.01f);   // /TEMP
        const float emit = accC[r] * emb;
        const float full = (q < 6) ? 1.f : (accL[r] * accR[r]);
        float o = emit * full;
        if (l == 0)         o *= eh[b * 30 + q];
        if (l == L_SEQ - 1) o *= eh[b * 30 + 15 + q];
        orow[q] = o;
      }
    }
  }
}

// ---------------------------------------------------------------------------
extern "C" void kernel_launch(void* const* d_in, const int* in_sizes, int n_in,
                              void* d_out, int out_size, void* d_ws, size_t ws_size,
                              hipStream_t stream) {
  (void)in_sizes; (void)n_in; (void)out_size; (void)ws_size;
  const float* in = (const float*)d_in[0];                 // (1,16,8192,84)
  const float* eh = (const float*)d_in[1];                 // (1,16,2,15)
  const float* ek = (const float*)d_in[2];                 // (1,15,15)
  const float* ee = (const float*)d_in[3];                 // (1,15,1,128)
  float* out = (float*)d_out;

  char* ws = (char*)d_ws;
  _Float16* wsW   = (_Float16*)(ws);          // 16*128 halves = 4096 B
  _Float16* wsBsm = (_Float16*)(ws + 4096);   // 16*32  halves = 1024 B
  _Float16* wsL   = (_Float16*)(ws + 5120);   // 16*64  halves = 2048 B
  _Float16* wsR   = (_Float16*)(ws + 7168);   // 2048 B
  float*    wsKq  = (float*)   (ws + 9216);   // 64 B

  genehmm_prep_kernel<<<dim3(1), dim3(256), 0, stream>>>(ek, ee, wsW, wsKq,
                                                         wsBsm, wsL, wsR);
  genehmm_emitter_kernel<<<dim3(1024), dim3(128), 0, stream>>>(
      in, eh, wsW, wsKq, wsBsm, wsL, wsR, out);
}
